// SchNet_15023795601941
// MI455X (gfx1250) — matrix operand order repaired and checked by reading, not compiled, not run on hardware
//
#include <hip/hip_runtime.h>
#include <hip/hip_bf16.h>

typedef __attribute__((ext_vector_type(16))) _Float16 v16h;
typedef __attribute__((ext_vector_type(8)))  float    v8f;
typedef __attribute__((ext_vector_type(4)))  int      v4i;

#define LN2F 0.69314718056f

// ---------------------------------------------------------------- async copy
#if defined(__HIP_DEVICE_COMPILE__) && __has_builtin(__builtin_amdgcn_global_load_async_to_lds_b128)
#define HAVE_ASYNC_LDS 1
#else
#define HAVE_ASYNC_LDS 0
#endif

__device__ __forceinline__ void copy16_to_lds(const _Float16* g, _Float16* l) {
#if HAVE_ASYNC_LDS
    __builtin_amdgcn_global_load_async_to_lds_b128(
        (__attribute__((address_space(1))) v4i*)(g),
        (__attribute__((address_space(3))) v4i*)(l), 0, 0);
#else
    *(int4*)l = *(const int4*)g;
#endif
}

__device__ __forceinline__ void copy_wait() {
#if HAVE_ASYNC_LDS
    asm volatile("s_wait_asynccnt 0" ::: "memory");
#endif
}

// ---------------------------------------------------------------- math utils
__device__ __forceinline__ float ssp_f(float x) {
    float ax = fabsf(x);
    return fmaxf(x, 0.0f) + __logf(1.0f + __expf(-ax)) - LN2F;
}

__device__ __forceinline__ v8f wmma16(v16h a, v16h b, v8f c) {
    return __builtin_amdgcn_wmma_f32_16x16x32_f16(
        false, a, false, b, (short)0, c, false, false);
}

// A fragment: 16x32 f16 tile at (mtile, ktile) from row-major [M][lda] halves.
__device__ __forceinline__ v16h load_frag_a(const _Float16* A, int lda,
                                            int mtile, int ktile, int lane) {
    int r  = lane & 15;
    int hi = lane >> 4;
    const _Float16* p = A + (mtile * 16 + r) * lda + ktile * 32;
    v16h a;
#pragma unroll
    for (int e = 0; e < 8; ++e) {
        a[e]     = p[hi * 16 + e];
        a[e + 8] = p[hi * 16 + 8 + e];
    }
    return a;
}

// B fragment from pre-packed (fragment-native) weight layout:
// frag (kt, nt) for lane L lives at [((kt*8 + nt)*32 + L) * 16 .. +15]
__device__ __forceinline__ v16h load_frag_bp(const _Float16* P, int kt, int nt,
                                             int lane) {
    return *(const v16h*)(P + (((kt << 3) + nt) * 32 + lane) * 16);
}

// ---------------------------------------------------------------- packers
// pack [128][128] f32 weight into WMMA-B fragment order (f16)
__global__ void k_pack_b128(const float* __restrict__ src,
                            _Float16* __restrict__ dst) {
    int idx = blockIdx.x * blockDim.x + threadIdx.x;   // 16384
    if (idx >= 128 * 128) return;
    int e    = idx & 15;
    int lane = (idx >> 4) & 31;
    int nt   = (idx >> 9) & 7;
    int kt   = idx >> 12;
    int K    = kt * 32 + (lane >> 4) * 16 + e;
    int col  = nt * 16 + (lane & 15);
    dst[idx] = (_Float16)src[K * 128 + col];
}

// pack fw1 [25][128] into single-ktile fragment order, zero-padding K to 32
__global__ void k_pack_b_fw1(const float* __restrict__ src,
                             _Float16* __restrict__ dst) {
    int idx = blockIdx.x * blockDim.x + threadIdx.x;   // 4096
    if (idx >= 8 * 32 * 16) return;
    int e    = idx & 15;
    int lane = (idx >> 4) & 31;
    int nt   = idx >> 9;
    int K    = (lane >> 4) * 16 + e;
    int col  = nt * 16 + (lane & 15);
    dst[idx] = (K < 25) ? (_Float16)src[K * 128 + col] : (_Float16)0.0f;
}

__global__ void k_embed(const int* __restrict__ Z,
                        const float* __restrict__ emb,
                        float* __restrict__ x, _Float16* __restrict__ x16) {
    int atom = blockIdx.x;      // B*A = 8192
    int f    = threadIdx.x;     // 128
    float v  = emb[Z[atom] * 128 + f];
    x[atom * 128 + f]   = v;
    x16[atom * 128 + f] = (_Float16)v;
}

// ------------------------------------------------ g16 = x16 @ in2f_w (f16 out)
__global__ __launch_bounds__(128)
void k_rowgemm_in2f(const _Float16* __restrict__ x16,
                    const _Float16* __restrict__ wp,
                    _Float16* __restrict__ g16) {
    __shared__ alignas(16) _Float16 ash[64 * 128];   // 16 KB
    const int tid = threadIdx.x, lane = tid & 31, wave = tid >> 5;
    const int row0 = blockIdx.x * 64;

    const _Float16* src = x16 + row0 * 128;
    for (int ofs = tid * 8; ofs < 64 * 128; ofs += 128 * 8)
        copy16_to_lds(src + ofs, ash + ofs);

    v16h bf[2][4];
#pragma unroll
    for (int t = 0; t < 2; ++t)
#pragma unroll
        for (int kt = 0; kt < 4; ++kt)
            bf[t][kt] = load_frag_bp(wp, kt, wave * 2 + t, lane);

    copy_wait();
    __syncthreads();

    for (int mt = 0; mt < 4; ++mt) {
        v16h a[4];
#pragma unroll
        for (int kt = 0; kt < 4; ++kt)
            a[kt] = load_frag_a(ash, 128, mt, kt, lane);
        int rb = row0 + mt * 16 + ((lane >> 4) << 3);
#pragma unroll
        for (int t = 0; t < 2; ++t) {
            v8f acc = {};
#pragma unroll
            for (int kt = 0; kt < 4; ++kt) acc = wmma16(a[kt], bf[t][kt], acc);
            int col = (wave * 2 + t) * 16 + (lane & 15);
#pragma unroll
            for (int r = 0; r < 8; ++r)
                g16[(rb + r) * 128 + col] = (_Float16)acc[r];
        }
    }
}

// ------------------------------------------------------------- fused CFConv
__global__ __launch_bounds__(128)
void k_cfconv(const float* __restrict__ pos,
              const float* __restrict__ cell,
              const float* __restrict__ cell_off,
              const int*   __restrict__ nbrs,
              const float* __restrict__ nmask,
              const _Float16* __restrict__ fw1p,   // packed, 1 k-tile
              const float* __restrict__ fb1,
              const _Float16* __restrict__ fw2p,   // packed, 4 k-tiles
              const float* __restrict__ fb2,
              const _Float16* __restrict__ g16,
              _Float16* __restrict__ y16) {
    __shared__ alignas(16) _Float16 fsh[64 * 32];    //  4 KB RBF (K padded)
    __shared__ alignas(16) _Float16 hsh[64 * 128];   // 16 KB hidden after ssp
    __shared__ alignas(16) _Float16 gsh[64 * 128];   // 16 KB gathered g rows
    __shared__ float dist_sh[64];
    __shared__ float cw_sh[64];
    __shared__ int   nbr_sh[64];

    const int tid  = threadIdx.x, lane = tid & 31, wave = tid >> 5;
    const int atom = blockIdx.x;          // b*1024 + a
    const int b    = atom >> 10;
    const int base = atom * 64;

    // register-resident B fragments (two b128 loads each; issue early)
    v16h bf1[2];
    v16h bf2[2][4];
#pragma unroll
    for (int t = 0; t < 2; ++t) {
        bf1[t] = load_frag_bp(fw1p, 0, wave * 2 + t, lane);
#pragma unroll
        for (int kt = 0; kt < 4; ++kt)
            bf2[t][kt] = load_frag_bp(fw2p, kt, wave * 2 + t, lane);
    }

    if (tid < 64) {
        int   n = tid;
        int   j = nbrs[base + n];
        float m = nmask[base + n];
        nbr_sh[n] = j;
        const float* pi = pos + atom * 3;
        const float* pj = pos + ((b << 10) + j) * 3;
        const float* co = cell_off + (base + n) * 3;
        const float* cl = cell + b * 9;
        float ox = co[0] * cl[0] + co[1] * cl[3] + co[2] * cl[6];
        float oy = co[0] * cl[1] + co[1] * cl[4] + co[2] * cl[7];
        float oz = co[0] * cl[2] + co[1] * cl[5] + co[2] * cl[8];
        float dx = pj[0] - pi[0] + ox;
        float dy = pj[1] - pi[1] + oy;
        float dz = pj[2] - pi[2] + oz;
        float sq = dx * dx + dy * dy + dz * dz;
        float d  = sqrtf(fmaxf(sq, 1e-12f)) * m;
        dist_sh[n] = d;
        cw_sh[n]   = (d <= 5.0f ? 1.0f : 0.0f) * m;
    }
    __syncthreads();

    // GaussianSmearing(0, 5, 25), K padded to 32 with zeros
    const float width = 5.0f / 24.0f;
    const float coeff = -0.5f / (width * width);
    for (int idx = tid; idx < 64 * 32; idx += 128) {
        int n = idx >> 5, k = idx & 31;
        float v = 0.0f;
        if (k < 25) {
            float t = dist_sh[n] - (float)k * width;
            v = __expf(coeff * t * t);
        }
        fsh[idx] = (_Float16)v;
    }
    // async-gather the 64 neighbor rows of g16 into LDS (16B chunks)
    for (int idx = tid; idx < 64 * 16; idx += 128) {
        int n = idx >> 4, c = idx & 15;
        copy16_to_lds(g16 + ((b << 10) + nbr_sh[n]) * 128 + c * 8,
                      gsh + n * 128 + c * 8);
    }
    copy_wait();
    __syncthreads();

    // GEMM1: H = ssp(fsh[64x32] @ fw1[32x128] + fb1)
    for (int mt = 0; mt < 4; ++mt) {
        v16h a  = load_frag_a(fsh, 32, mt, 0, lane);
        int  rl = mt * 16 + ((lane >> 4) << 3);
#pragma unroll
        for (int t = 0; t < 2; ++t) {
            v8f z = {};
            v8f acc = wmma16(a, bf1[t], z);
            int   col = (wave * 2 + t) * 16 + (lane & 15);
            float bb  = fb1[col];
#pragma unroll
            for (int r = 0; r < 8; ++r)
                hsh[(rl + r) * 128 + col] = (_Float16)ssp_f(acc[r] + bb);
        }
    }
    __syncthreads();

    // GEMM2: W = H @ fw2 + fb2 (gated), y[f] = sum_n gsh[n][f] * W[n][f]
    const int col0 = (wave * 2) * 16 + (lane & 15);
    float ypart[2] = {0.0f, 0.0f};
    const float bb2[2] = {fb2[col0], fb2[col0 + 16]};
    for (int mt = 0; mt < 4; ++mt) {
        v16h a[4];
#pragma unroll
        for (int kt = 0; kt < 4; ++kt)
            a[kt] = load_frag_a(hsh, 128, mt, kt, lane);
        int rl = mt * 16 + ((lane >> 4) << 3);
#pragma unroll
        for (int t = 0; t < 2; ++t) {
            v8f acc = {};
#pragma unroll
            for (int kt = 0; kt < 4; ++kt) acc = wmma16(a[kt], bf2[t][kt], acc);
            int col = col0 + t * 16;
#pragma unroll
            for (int r = 0; r < 8; ++r) {
                int   n = rl + r;
                float w = (acc[r] + bb2[t]) * cw_sh[n];
                ypart[t] += w * (float)gsh[n * 128 + col];
            }
        }
    }
#pragma unroll
    for (int t = 0; t < 2; ++t) {
        float yp = ypart[t] + __shfl_xor(ypart[t], 16, 32);
        if (lane < 16) y16[atom * 128 + col0 + t * 16] = (_Float16)yp;
    }
}

// ------------------- y2 = ssp(y@f2out + b); x += y2@dense + b; refresh x16
__global__ __launch_bounds__(128)
void k_post(const _Float16* __restrict__ y16,
            const _Float16* __restrict__ w1p, const float* __restrict__ b1,
            const _Float16* __restrict__ w2p, const float* __restrict__ b2,
            float* __restrict__ x, _Float16* __restrict__ x16) {
    __shared__ alignas(16) _Float16 ash[64 * 128];    // 16 KB
    __shared__ alignas(16) _Float16 y2sh[64 * 128];   // 16 KB
    const int tid = threadIdx.x, lane = tid & 31, wave = tid >> 5;
    const int row0 = blockIdx.x * 64;

    const _Float16* src = y16 + row0 * 128;
    for (int ofs = tid * 8; ofs < 64 * 128; ofs += 128 * 8)
        copy16_to_lds(src + ofs, ash + ofs);

    v16h bf[2][4];
#pragma unroll
    for (int t = 0; t < 2; ++t)
#pragma unroll
        for (int kt = 0; kt < 4; ++kt)
            bf[t][kt] = load_frag_bp(w1p, kt, wave * 2 + t, lane);

    copy_wait();
    __syncthreads();

    for (int mt = 0; mt < 4; ++mt) {
        v16h a[4];
#pragma unroll
        for (int kt = 0; kt < 4; ++kt)
            a[kt] = load_frag_a(ash, 128, mt, kt, lane);
        int rl = mt * 16 + ((lane >> 4) << 3);
#pragma unroll
        for (int t = 0; t < 2; ++t) {
            v8f acc = {};
#pragma unroll
            for (int kt = 0; kt < 4; ++kt) acc = wmma16(a[kt], bf[t][kt], acc);
            int   col = (wave * 2 + t) * 16 + (lane & 15);
            float bb  = b1[col];
#pragma unroll
            for (int r = 0; r < 8; ++r)
                y2sh[(rl + r) * 128 + col] = (_Float16)ssp_f(acc[r] + bb);
        }
    }

    // reload B fragments with the dense weights
#pragma unroll
    for (int t = 0; t < 2; ++t)
#pragma unroll
        for (int kt = 0; kt < 4; ++kt)
            bf[t][kt] = load_frag_bp(w2p, kt, wave * 2 + t, lane);
    __syncthreads();

    for (int mt = 0; mt < 4; ++mt) {
        v16h a[4];
#pragma unroll
        for (int kt = 0; kt < 4; ++kt)
            a[kt] = load_frag_a(y2sh, 128, mt, kt, lane);
        int rb = row0 + mt * 16 + ((lane >> 4) << 3);
#pragma unroll
        for (int t = 0; t < 2; ++t) {
            v8f acc = {};
#pragma unroll
            for (int kt = 0; kt < 4; ++kt) acc = wmma16(a[kt], bf[t][kt], acc);
            int   col = (wave * 2 + t) * 16 + (lane & 15);
            float bb  = b2[col];
#pragma unroll
            for (int r = 0; r < 8; ++r) {
                int   idx = (rb + r) * 128 + col;
                float nv  = x[idx] + acc[r] + bb;
                x[idx]    = nv;
                x16[idx]  = (_Float16)nv;
            }
        }
    }
}

// ---------------------------------------------------------------------------

extern "C" void kernel_launch(void* const* d_in, const int* in_sizes, int n_in,
                              void* d_out, int out_size, void* d_ws, size_t ws_size,
                              hipStream_t stream) {
    (void)in_sizes; (void)n_in; (void)out_size; (void)ws_size;

    const int*   Z        = (const int*)  d_in[0];
    const float* pos      = (const float*)d_in[1];
    const float* cell     = (const float*)d_in[2];
    const float* cell_off = (const float*)d_in[3];
    const int*   nbrs     = (const int*)  d_in[4];
    const float* nmask    = (const float*)d_in[5];
    const float* emb      = (const float*)d_in[6];
    const float* fw1      = (const float*)d_in[7];
    const float* fb1      = (const float*)d_in[8];
    const float* fw2      = (const float*)d_in[9];
    const float* fb2      = (const float*)d_in[10];
    const float* in2f_w   = (const float*)d_in[11];
    const float* f2out_w  = (const float*)d_in[12];
    const float* f2out_b  = (const float*)d_in[13];
    const float* dense_w  = (const float*)d_in[14];
    const float* dense_b  = (const float*)d_in[15];

    const int BA = 8 * 1024;           // 8192 atoms
    float* x = (float*)d_out;          // x (f32) lives in d_out

    // workspace layout (f16 mirrors + packed f16 weights)
    char* ws = (char*)d_ws;
    _Float16* x16 = (_Float16*)ws;  ws += (size_t)BA * 128 * sizeof(_Float16);
    _Float16* g16 = (_Float16*)ws;  ws += (size_t)BA * 128 * sizeof(_Float16);
    _Float16* y16 = (_Float16*)ws;  ws += (size_t)BA * 128 * sizeof(_Float16);
    _Float16* w16 = (_Float16*)ws;

    const int SZ_FW1P = 8 * 32 * 16;   // packed fw1 (1 k-tile)
    const int SZ_SQ   = 128 * 128;     // packed square weights (4 k-tiles)
    const int PER     = SZ_FW1P + 4 * SZ_SQ;

    for (int i = 0; i < 3; ++i) {
        _Float16* s = w16 + (size_t)i * PER;
        k_pack_b_fw1<<<(SZ_FW1P + 255) / 256, 256, 0, stream>>>(fw1 + i * 25 * 128, s);
        s += SZ_FW1P;
        k_pack_b128<<<(SZ_SQ + 255) / 256, 256, 0, stream>>>(fw2     + i * SZ_SQ, s);
        s += SZ_SQ;
        k_pack_b128<<<(SZ_SQ + 255) / 256, 256, 0, stream>>>(in2f_w  + i * SZ_SQ, s);
        s += SZ_SQ;
        k_pack_b128<<<(SZ_SQ + 255) / 256, 256, 0, stream>>>(f2out_w + i * SZ_SQ, s);
        s += SZ_SQ;
        k_pack_b128<<<(SZ_SQ + 255) / 256, 256, 0, stream>>>(dense_w + i * SZ_SQ, s);
    }

    k_embed<<<BA, 128, 0, stream>>>(Z, emb, x, x16);

    for (int i = 0; i < 3; ++i) {
        const _Float16* slab   = w16 + (size_t)i * PER;
        const _Float16* fw1p   = slab;
        const _Float16* fw2p   = slab + SZ_FW1P;
        const _Float16* in2fp  = slab + SZ_FW1P + SZ_SQ;
        const _Float16* f2op   = slab + SZ_FW1P + 2 * SZ_SQ;
        const _Float16* dnsp   = slab + SZ_FW1P + 3 * SZ_SQ;

        k_rowgemm_in2f<<<BA / 64, 128, 0, stream>>>(x16, in2fp, g16);

        k_cfconv<<<BA, 128, 0, stream>>>(pos, cell, cell_off, nbrs, nmask,
                                         fw1p, fb1 + i * 128,
                                         fw2p, fb2 + i * 128,
                                         g16, y16);

        k_post<<<BA / 64, 128, 0, stream>>>(y16, f2op, f2out_b + i * 128,
                                            dnsp, dense_b + i * 128, x, x16);
    }
}
